// CustomGPT2MultiHeadAttention_66108136620059
// MI455X (gfx1250) — matrix operand order repaired and checked by reading, not compiled, not run on hardware
//
#include <hip/hip_runtime.h>
#include <hip/hip_bf16.h>

typedef __attribute__((ext_vector_type(16))) __bf16 v16bf;
typedef __attribute__((ext_vector_type(8)))  float  v8f;

#define DEV __device__ __forceinline__

// ---- bf16 helpers (round-to-nearest-even), storage kept as u16/u32 ----
DEV unsigned short f32_bf16(float f) {
  unsigned u = __builtin_bit_cast(unsigned, f);
  unsigned r = 0x7FFFu + ((u >> 16) & 1u);
  return (unsigned short)((u + r) >> 16);
}
DEV unsigned pack2(float lo, float hi) {
  return (unsigned)f32_bf16(lo) | ((unsigned)f32_bf16(hi) << 16);
}

union AF { v16bf v; unsigned u[8]; };

DEV v8f wmma_bf16(const AF& a, const AF& b, v8f c) {
  // D = A(16x32 bf16) * B(32x16 bf16) + C(16x16 f32)
  return __builtin_amdgcn_wmma_f32_16x16x32_bf16(false, a.v, false, b.v,
                                                 (short)0, c, false, false);
}

// A-fragment pair index within a 32-K tile: v -> v+8h
DEV int apair(int v, int hl) { return v + 8 * hl; }

// ====================================================================
// Stage 1: QKV GEMM  X[4096,1024] * W[1024,3072] + b -> Q,K (B,H,S,64) ; V^T (B,H,64,S)  (bf16)
// ====================================================================
__global__ __launch_bounds__(256) void qkv_kernel(
    const float* __restrict__ X, const float* __restrict__ W,
    const float* __restrict__ bias,
    unsigned short* __restrict__ Qo, unsigned short* __restrict__ Ko,
    unsigned short* __restrict__ VTo)
{
  __shared__ unsigned As[128][17];  // [m][kpair]
  __shared__ unsigned Bs[128][17];  // [n][kpair]
  const int tid = threadIdx.x;
  const int lane = tid & 31, wave = tid >> 5;
  const int hl = lane >> 4, ln = lane & 15;
  const int m0 = blockIdx.y * 128, n0 = blockIdx.x * 128;
  const int wm = (wave & 3) * 32, wn = (wave >> 2) * 64;

  v8f acc[2][4];
#pragma unroll
  for (int i = 0; i < 2; ++i)
#pragma unroll
    for (int j = 0; j < 4; ++j) acc[i][j] = (v8f){0,0,0,0,0,0,0,0};

  for (int k0 = 0; k0 < 1024; k0 += 32) {
    if (k0 + 32 < 1024) {
      __builtin_prefetch(X + (m0 + (tid >> 1)) * 1024 + (k0 + 32) + (tid & 1) * 16, 0, 0);
      __builtin_prefetch(W + (k0 + 32 + (tid >> 5)) * 3072 + n0 + (tid & 31) * 4, 0, 0);
    }
#pragma unroll
    for (int i = 0; i < 4; ++i) {               // A tile: 128x32 f32 -> bf16 pairs
      int idx = tid + i * 256;
      int row = idx >> 3, cs = idx & 7;
      const float4 f = *(const float4*)(X + (m0 + row) * 1024 + k0 + cs * 4);
      As[row][cs * 2]     = pack2(f.x, f.y);
      As[row][cs * 2 + 1] = pack2(f.z, f.w);
    }
#pragma unroll
    for (int i = 0; i < 2; ++i) {               // B tile: 32x128, transposed to [n][kpair]
      int idx = tid + i * 256;
      int kp = idx >> 5, ng = idx & 31;
      const float4 r0 = *(const float4*)(W + (k0 + 2 * kp) * 3072 + n0 + ng * 4);
      const float4 r1 = *(const float4*)(W + (k0 + 2 * kp + 1) * 3072 + n0 + ng * 4);
      Bs[ng * 4 + 0][kp] = pack2(r0.x, r1.x);
      Bs[ng * 4 + 1][kp] = pack2(r0.y, r1.y);
      Bs[ng * 4 + 2][kp] = pack2(r0.z, r1.z);
      Bs[ng * 4 + 3][kp] = pack2(r0.w, r1.w);
    }
    __syncthreads();
    AF a[2], b[4];
#pragma unroll
    for (int mt = 0; mt < 2; ++mt)
#pragma unroll
      for (int v = 0; v < 8; ++v)
        a[mt].u[v] = As[wm + mt * 16 + ln][apair(v, hl)];
#pragma unroll
    for (int nt = 0; nt < 4; ++nt)
#pragma unroll
      for (int v = 0; v < 8; ++v)
        b[nt].u[v] = Bs[wn + nt * 16 + ln][8 * hl + v];
#pragma unroll
    for (int mt = 0; mt < 2; ++mt)
#pragma unroll
      for (int nt = 0; nt < 4; ++nt)
        acc[mt][nt] = wmma_bf16(a[mt], b[nt], acc[mt][nt]);
    __syncthreads();
  }

  // Scatter to Q/K ([B,H,S,64]) and V^T ([B,H,64,S]) as bf16, adding bias.
#pragma unroll
  for (int mt = 0; mt < 2; ++mt)
#pragma unroll
    for (int nt = 0; nt < 4; ++nt) {
      int col = n0 + wn + nt * 16 + ln;      // 0..3071
      int which = col >> 10;                 // 0=q 1=k 2=v (uniform per fragment)
      int dcol = col & 1023;
      int head = dcol >> 6, hdc = dcol & 63;
      float bv = bias[col];
      int rowb = m0 + wm + mt * 16 + 8 * hl;
#pragma unroll
      for (int r = 0; r < 8; ++r) {
        int grow = rowb + r;
        int bb = grow >> 11, ss = grow & 2047;
        unsigned short h16 = f32_bf16(acc[mt][nt][r] + bv);
        if (which == 0)
          Qo[((bb * 16 + head) * 2048 + ss) * 64 + hdc] = h16;
        else if (which == 1)
          Ko[((bb * 16 + head) * 2048 + ss) * 64 + hdc] = h16;
        else
          VTo[((bb * 16 + head) * 64 + hdc) * 2048 + ss] = h16;
      }
    }
}

// ====================================================================
// Stage 2: causal flash attention per (b,h). One wave per 16-row Q strip.
// ====================================================================
__global__ __launch_bounds__(256) void attn_kernel(
    const unsigned short* __restrict__ Qw, const unsigned short* __restrict__ Kw,
    const unsigned short* __restrict__ VTw, unsigned short* __restrict__ Ow)
{
  __shared__ unsigned short Ps[8][16][66];   // wave-private P scratch (padded rows)
  const int tid = threadIdx.x, lane = tid & 31, wave = tid >> 5;
  const int hl = lane >> 4, ln = lane & 15;
  const int bh = blockIdx.y;
  const int qb = blockIdx.x * 128 + wave * 16;
  const unsigned* Qu = (const unsigned*)Qw + bh * 2048 * 32;
  const unsigned* Ku = (const unsigned*)Kw + bh * 2048 * 32;
  const unsigned* Vu = (const unsigned*)VTw + bh * 64 * 1024;  // [64][1024] uint pairs
  unsigned short* Oh = Ow + bh * 2048 * 64;

  AF q[2];
#pragma unroll
  for (int kc = 0; kc < 2; ++kc)
#pragma unroll
    for (int v = 0; v < 8; ++v)
      q[kc].u[v] = Qu[(qb + ln) * 32 + kc * 16 + apair(v, hl)];

  v8f o[4];
#pragma unroll
  for (int t = 0; t < 4; ++t) o[t] = (v8f){0,0,0,0,0,0,0,0};
  float M[8], L[8];
#pragma unroll
  for (int r = 0; r < 8; ++r) { M[r] = -1e30f; L[r] = 0.0f; }

  for (int kv = 0; kv <= qb + 15; kv += 64) {
    // S = Q * K^T  (4 N-tiles of 16 kv cols, K-dim 64 split into 2x32)
    v8f s[4];
#pragma unroll
    for (int t = 0; t < 4; ++t) {
      AF kf0, kf1;
      int kr = kv + t * 16 + ln;
#pragma unroll
      for (int v = 0; v < 8; ++v) {
        kf0.u[v] = Ku[kr * 32 + 8 * hl + v];
        kf1.u[v] = Ku[kr * 32 + 16 + 8 * hl + v];
      }
      v8f z = (v8f){0,0,0,0,0,0,0,0};
      z = wmma_bf16(q[0], kf0, z);
      s[t] = wmma_bf16(q[1], kf1, z);
    }
    // scale + causal mask
#pragma unroll
    for (int t = 0; t < 4; ++t) {
      int kg = kv + t * 16 + ln;
#pragma unroll
      for (int r = 0; r < 8; ++r) {
        int qg = qb + 8 * hl + r;
        float val = s[t][r] * 0.125f;             // 1/sqrt(64)
        s[t][r] = (kg > qg) ? -10000.0f : val;
      }
    }
    // per-row max (rows live per-VGPR; reduce over 16 lanes of the half)
    float mn[8];
#pragma unroll
    for (int r = 0; r < 8; ++r)
      mn[r] = fmaxf(fmaxf(s[0][r], s[1][r]), fmaxf(s[2][r], s[3][r]));
    for (int off = 1; off < 16; off <<= 1)
#pragma unroll
      for (int r = 0; r < 8; ++r)
        mn[r] = fmaxf(mn[r], __shfl_xor(mn[r], off, 32));
    float rs[8];
#pragma unroll
    for (int r = 0; r < 8; ++r) {
      float Mn = fmaxf(M[r], mn[r]);
      rs[r] = __expf(M[r] - Mn);
      M[r] = Mn;
      L[r] *= rs[r];
    }
#pragma unroll
    for (int t = 0; t < 4; ++t)
#pragma unroll
      for (int r = 0; r < 8; ++r) o[t][r] *= rs[r];
    // P = exp(S - M), row sums
    float sum[8];
#pragma unroll
    for (int r = 0; r < 8; ++r) sum[r] = 0.0f;
#pragma unroll
    for (int t = 0; t < 4; ++t)
#pragma unroll
      for (int r = 0; r < 8; ++r) {
        float p = __expf(s[t][r] - M[r]);
        s[t][r] = p;
        sum[r] += p;
      }
    for (int off = 1; off < 16; off <<= 1)
#pragma unroll
      for (int r = 0; r < 8; ++r)
        sum[r] += __shfl_xor(sum[r], off, 32);
#pragma unroll
    for (int r = 0; r < 8; ++r) L[r] += sum[r];

    // C-layout -> A-layout for P via wave-private LDS (DS ops in-order per wave)
#pragma unroll
    for (int t = 0; t < 4; ++t)
#pragma unroll
      for (int r = 0; r < 8; ++r)
        Ps[wave][8 * hl + r][t * 16 + ln] = f32_bf16(s[t][r]);
    __asm__ volatile("" ::: "memory");
    AF pf[2];
    const unsigned* prow = (const unsigned*)&Ps[wave][ln][0];
#pragma unroll
    for (int kc = 0; kc < 2; ++kc)
#pragma unroll
      for (int v = 0; v < 8; ++v)
        pf[kc].u[v] = prow[kc * 16 + apair(v, hl)];
    __asm__ volatile("" ::: "memory");

    // O += P * V  (V^T layout gives contiguous dword B-fragment loads)
#pragma unroll
    for (int t = 0; t < 4; ++t) {
      AF vf0, vf1;
      int hr = t * 16 + ln;
#pragma unroll
      for (int v = 0; v < 8; ++v) {
        vf0.u[v] = Vu[hr * 1024 + (kv >> 1) + 8 * hl + v];
        vf1.u[v] = Vu[hr * 1024 + (kv >> 1) + 16 + 8 * hl + v];
      }
      o[t] = wmma_bf16(pf[0], vf0, o[t]);
      o[t] = wmma_bf16(pf[1], vf1, o[t]);
    }
  }

  // normalize and store O as bf16 [B,H,S,64]
  float inv[8];
#pragma unroll
  for (int r = 0; r < 8; ++r) inv[r] = 1.0f / L[r];
#pragma unroll
  for (int t = 0; t < 4; ++t)
#pragma unroll
    for (int r = 0; r < 8; ++r)
      Oh[(qb + 8 * hl + r) * 64 + t * 16 + ln] = f32_bf16(o[t][r] * inv[r]);
}

// ====================================================================
// Stage 3: out = O[4096,1024](bf16, [B,H,S,64]) * Wp[1024,1024] + b  (f32 out)
// ====================================================================
__global__ __launch_bounds__(256) void proj_kernel(
    const unsigned* __restrict__ Au,        // O as uint pairs: [B*H][2048][32]
    const float* __restrict__ W, const float* __restrict__ bias,
    float* __restrict__ out)
{
  __shared__ unsigned As[128][17];
  __shared__ unsigned Bs[128][17];
  const int tid = threadIdx.x;
  const int lane = tid & 31, wave = tid >> 5;
  const int hl = lane >> 4, ln = lane & 15;
  const int m0 = blockIdx.y * 128, n0 = blockIdx.x * 128;
  const int wm = (wave & 3) * 32, wn = (wave >> 2) * 64;

  v8f acc[2][4];
#pragma unroll
  for (int i = 0; i < 2; ++i)
#pragma unroll
    for (int j = 0; j < 4; ++j) acc[i][j] = (v8f){0,0,0,0,0,0,0,0};

  for (int k0 = 0; k0 < 1024; k0 += 32) {
    const int head = k0 >> 6;
    const int kh = (k0 & 63) >> 1;   // uint offset inside the head's 32-uint row slice
    if (k0 + 32 < 1024)
      __builtin_prefetch(W + (k0 + 32 + (tid >> 5)) * 1024 + n0 + (tid & 31) * 4, 0, 0);
#pragma unroll
    for (int i = 0; i < 2; ++i) {              // A tile: already bf16 pairs, uint4 loads
      int idx = tid + i * 256;
      int row = idx >> 2, pq = (idx & 3) * 4;
      int grow = m0 + row, bb = grow >> 11, ss = grow & 2047;
      const uint4 f = *(const uint4*)(Au + ((bb * 16 + head) * 2048 + ss) * 32 + kh + pq);
      As[row][pq] = f.x; As[row][pq + 1] = f.y;
      As[row][pq + 2] = f.z; As[row][pq + 3] = f.w;
    }
#pragma unroll
    for (int i = 0; i < 2; ++i) {              // B tile f32 -> bf16, transposed
      int idx = tid + i * 256;
      int kp = idx >> 5, ng = idx & 31;
      const float4 r0 = *(const float4*)(W + (k0 + 2 * kp) * 1024 + n0 + ng * 4);
      const float4 r1 = *(const float4*)(W + (k0 + 2 * kp + 1) * 1024 + n0 + ng * 4);
      Bs[ng * 4 + 0][kp] = pack2(r0.x, r1.x);
      Bs[ng * 4 + 1][kp] = pack2(r0.y, r1.y);
      Bs[ng * 4 + 2][kp] = pack2(r0.z, r1.z);
      Bs[ng * 4 + 3][kp] = pack2(r0.w, r1.w);
    }
    __syncthreads();
    AF a[2], b[4];
#pragma unroll
    for (int mt = 0; mt < 2; ++mt)
#pragma unroll
      for (int v = 0; v < 8; ++v)
        a[mt].u[v] = As[wm + mt * 16 + ln][apair(v, hl)];
#pragma unroll
    for (int nt = 0; nt < 4; ++nt)
#pragma unroll
      for (int v = 0; v < 8; ++v)
        b[nt].u[v] = Bs[wn + nt * 16 + ln][8 * hl + v];
#pragma unroll
    for (int mt = 0; mt < 2; ++mt)
#pragma unroll
      for (int nt = 0; nt < 4; ++nt)
        acc[mt][nt] = wmma_bf16(a[mt], b[nt], acc[mt][nt]);
    __syncthreads();
  }

#pragma unroll
  for (int mt = 0; mt < 2; ++mt)
#pragma unroll
    for (int nt = 0; nt < 4; ++nt) {
      int col = n0 + wn + nt * 16 + ln;
      float bv = bias[col];
      int rowb = m0 + wm + mt * 16 + 8 * hl;
#pragma unroll
      for (int r = 0; r < 8; ++r)
        out[(size_t)(rowb + r) * 1024 + col] = acc[mt][nt][r] + bv;
    }
}

// ====================================================================
extern "C" void kernel_launch(void* const* d_in, const int* in_sizes, int n_in,
                              void* d_out, int out_size, void* d_ws, size_t ws_size,
                              hipStream_t stream) {
  (void)in_sizes; (void)n_in; (void)out_size; (void)ws_size;
  const float* X    = (const float*)d_in[0];   // [2,2048,1024]
  const float* Wqkv = (const float*)d_in[1];   // [1024,3072]
  const float* Bqkv = (const float*)d_in[2];   // [3072]
  const float* Wp   = (const float*)d_in[3];   // [1024,1024]
  const float* Bp   = (const float*)d_in[4];   // [1024]
  float* out = (float*)d_out;

  const size_t E = (size_t)2 * 16 * 2048 * 64; // 4M bf16 elements per tensor
  unsigned short* Q  = (unsigned short*)d_ws;
  unsigned short* K  = Q + E;
  unsigned short* VT = K + E;
  unsigned short* O  = VT + E;

  qkv_kernel <<<dim3(24, 32), 256, 0, stream>>>(X, Wqkv, Bqkv, Q, K, VT);
  attn_kernel<<<dim3(16, 32), 256, 0, stream>>>(Q, K, VT, O);
  proj_kernel<<<dim3(8, 32),  256, 0, stream>>>((const unsigned*)O, Wp, Bp, out);
}
